// MHAWithKVCache_66898410602879
// MI455X (gfx1250) — matrix operand order repaired and checked by reading, not compiled, not run on hardware
//
#include <hip/hip_runtime.h>
#include <hip/hip_bf16.h>
#include <math.h>

typedef __attribute__((ext_vector_type(16))) __bf16 v16bf;
typedef __attribute__((ext_vector_type(8)))  float  v8f;
typedef __attribute__((ext_vector_type(4)))  float  fx4;

#define B_   16
#define SN_  4
#define D_   2048
#define H_   16
#define HD_  128
#define SP_  4096
#define ST_  4100
#define ROWS (B_*SN_)                      // 64
#define INV_SCALE 0.08838834764831845f     // 1/sqrt(128)

#define KV_ELEMS ((size_t)B_*H_*ST_*HD_)   // 134,348,800 floats per output cache
#define K_OFF ((size_t)(ROWS*D_))          // 131072 floats (attention output first)
#define V_OFF (K_OFF + KV_ELEMS)

#define NCHUNK 4                           // s-range splits per (b,h)
#define NT_TILES 257                       // ceil(4100/16)
#define TILES_PER_CHUNK 65                 // 65,65,65,62
#define KSTRIDE 132                        // padded LDS row stride (floats)
#define PART_STRIDE 130                    // per (bh,chunk,r): m, l, acc[128]

static __device__ __forceinline__ v8f wmma_bf16(const v16bf& a, const v16bf& b, const v8f& c) {
    return __builtin_amdgcn_wmma_f32_16x16x32_bf16(false, a, false, b, (short)0, c, false, false);
}

// ---------------------------------------------------------------------------
// Kernel 1: fused Q/K/V projections  y = x @ W^T  (64x2048x2048 each)
// One wave = one 16x16 output tile, K-loop of 32 with v_wmma_f32_16x16x32_bf16.
// q -> workspace; k_new/v_new -> rows 4096..4099 of the output caches.
// ---------------------------------------------------------------------------
__global__ void qkv_proj_kernel(const float* __restrict__ x,
                                const float* __restrict__ Wq,
                                const float* __restrict__ Wk,
                                const float* __restrict__ Wv,
                                float* __restrict__ q_ws,
                                float* __restrict__ out)
{
    const int wave = blockIdx.x * (blockDim.x >> 5) + (threadIdx.x >> 5);
    const int lane = threadIdx.x & 31;
    const int half = lane >> 4;
    const int l15  = lane & 15;

    const int mat  = wave / 512;        // 0=q, 1=k, 2=v
    const int tile = wave % 512;
    const int mt   = tile >> 7;         // 0..3   (M tiles of 16 over 64 rows)
    const int nt   = tile & 127;        // 0..127 (N tiles of 16 over 2048 cols)

    const float* W = (mat == 0) ? Wq : (mat == 1) ? Wk : Wv;

    v8f c;
#pragma unroll
    for (int i = 0; i < 8; ++i) c[i] = 0.0f;

    const float* xrow = x + (size_t)(mt * 16 + l15) * D_;   // A row this lane supplies
    const float* wrow = W + (size_t)(nt * 16 + l15) * D_;   // B col = W row (K contiguous)

    for (int k0 = 0; k0 < D_; k0 += 32) {
        // A 16x32 bf16: half 0 -> K {0..7,16..23}, half 1 -> K {8..15,24..31}
        fx4 a0 = *(const fx4*)(xrow + k0 + 8 * half);
        fx4 a1 = *(const fx4*)(xrow + k0 + 8 * half + 4);
        fx4 a2 = *(const fx4*)(xrow + k0 + 16 + 8 * half);
        fx4 a3 = *(const fx4*)(xrow + k0 + 16 + 8 * half + 4);
        // B 32x16 bf16: half 0 -> K {0..7,16..23}, half 1 -> K {8..15,24..31}
        fx4 b0 = *(const fx4*)(wrow + k0 + 8 * half);
        fx4 b1 = *(const fx4*)(wrow + k0 + 8 * half + 4);
        fx4 b2 = *(const fx4*)(wrow + k0 + 16 + 8 * half);
        fx4 b3 = *(const fx4*)(wrow + k0 + 16 + 8 * half + 4);
        v16bf a, b;
#pragma unroll
        for (int e = 0; e < 4; ++e) {
            a[e]      = (__bf16)a0[e];  a[4 + e]  = (__bf16)a1[e];
            a[8 + e]  = (__bf16)a2[e];  a[12 + e] = (__bf16)a3[e];
            b[e]      = (__bf16)b0[e];  b[4 + e]  = (__bf16)b1[e];
            b[8 + e]  = (__bf16)b2[e];  b[12 + e] = (__bf16)b3[e];
        }
        c = wmma_bf16(a, b, c);
    }

    // D layout: VGPR v -> row v + 8*half, lanes 0..15 -> cols
#pragma unroll
    for (int v = 0; v < 8; ++v) {
        const int r  = mt * 16 + v + 8 * half;   // global row 0..63
        const int j  = nt * 16 + l15;            // global col 0..2047
        const int b_ = r >> 2, s_ = r & 3;
        const int h_ = j >> 7, hd = j & 127;
        const float val = c[v];
        if (mat == 0) {
            q_ws[(((size_t)(b_ * H_ + h_)) * SN_ + s_) * HD_ + hd] = val;
        } else {
            const size_t base = (mat == 1) ? K_OFF : V_OFF;
            out[base + (((size_t)(b_ * H_ + h_)) * ST_ + SP_ + s_) * HD_ + hd] = val;
        }
    }
}

// ---------------------------------------------------------------------------
// Kernel 2: fused (KV-concat copy + flash-decode attention partials).
// Grid = (b*h) x NCHUNK blocks; block = 4 waves; each wave streams s-tiles of
// 16 keys: coalesced NT row loads from k_cached/v_cached, NT store-through to
// the output cache (the concat!), K staged via padded LDS for WMMA B-frags,
// online softmax, VALU ctx accumulation. Un-normalized (m,l,acc) partials to ws.
// ---------------------------------------------------------------------------
__global__ void attn_partial_kernel(const float* __restrict__ q_ws,
                                    const float* __restrict__ kcache,
                                    const float* __restrict__ vcache,
                                    float* __restrict__ gout,
                                    float* __restrict__ part)
{
    __shared__ __align__(16) float k_lds[4][16 * KSTRIDE];   // per-wave K tile, padded
    __shared__ float red_m[4][4];
    __shared__ float red_l[4][4];
    __shared__ float red_acc[4][4][HD_];

    const int bh    = blockIdx.x >> 2;        // 0..255  (b*16+h)
    const int chunk = blockIdx.x & 3;         // 0..3
    const int wv    = threadIdx.x >> 5;       // 0..3
    const int lane  = threadIdx.x & 31;
    const int half  = lane >> 4;
    const int l15   = lane & 15;

    const size_t kin_base  = (size_t)bh * SP_ * HD_;                 // k_cached (b,h)
    float* kout = gout + K_OFF + (size_t)bh * ST_ * HD_;
    float* vout = gout + V_OFF + (size_t)bh * ST_ * HD_;

    // Q fragments (A matrices), rows >= 4 padded to zero
    v16bf afrag[4];
    {
        const float* qp = q_ws + (size_t)bh * SN_ * HD_;
#pragma unroll
        for (int kc4 = 0; kc4 < 4; ++kc4) {
            v16bf a;
#pragma unroll
            for (int e = 0; e < 16; ++e) a[e] = (__bf16)0.0f;
            if (l15 < SN_) {
                const float* qr = qp + l15 * HD_ + kc4 * 32;
#pragma unroll
                for (int e = 0; e < 8; ++e) {
                    a[e]     = (__bf16)qr[8 * half + e];
                    a[8 + e] = (__bf16)qr[16 + 8 * half + e];
                }
            }
            afrag[kc4] = a;
        }
    }

    float m_run[4], l_run[4], acc[4][4];
#pragma unroll
    for (int r = 0; r < 4; ++r) {
        m_run[r] = -1.0e30f;
        l_run[r] = 0.0f;
#pragma unroll
        for (int j = 0; j < 4; ++j) acc[r][j] = 0.0f;
    }

    const int t_begin = chunk * TILES_PER_CHUNK;
    const int t_end   = (t_begin + TILES_PER_CHUNK < NT_TILES) ? t_begin + TILES_PER_CHUNK
                                                               : NT_TILES;
    float* klds = &k_lds[wv][0];

    for (int t = t_begin + wv; t < t_end; t += 4) {
        // ---- cooperative K tile staging: coalesced row loads, NT store-through,
        // ---- LDS stash for the (strided) WMMA B-fragment gather.
#pragma unroll
        for (int si = 0; si < 16; ++si) {
            const int s = t * 16 + si;                 // uniform per wave
            fx4 kq = {0.0f, 0.0f, 0.0f, 0.0f};
            if (s < ST_) {
                const float* srow = (s < SP_)
                    ? (kcache + kin_base + (size_t)s * HD_)
                    : (kout + (size_t)s * HD_);        // new rows, written by proj
                kq = __builtin_nontemporal_load((const fx4*)(srow + 4 * lane));
                if (s < SP_)                           // the fused concat-copy
                    __builtin_nontemporal_store(kq, (fx4*)(kout + (size_t)s * HD_ + 4 * lane));
            }
            *(fx4*)(klds + si * KSTRIDE + 4 * lane) = kq;
        }
        __builtin_amdgcn_wave_barrier();

        // ---- scores tile via WMMA (B-fragments gathered from LDS)
        v8f c;
#pragma unroll
        for (int i = 0; i < 8; ++i) c[i] = 0.0f;
#pragma unroll
        for (int kc4 = 0; kc4 < 4; ++kc4) {
            const float* lp = klds + l15 * KSTRIDE + kc4 * 32 + 8 * half;
            fx4 b0 = *(const fx4*)(lp);
            fx4 b1 = *(const fx4*)(lp + 4);
            fx4 b2 = *(const fx4*)(lp + 16);
            fx4 b3 = *(const fx4*)(lp + 20);
            v16bf b;
#pragma unroll
            for (int e = 0; e < 4; ++e) {
                b[e]      = (__bf16)b0[e];  b[4 + e]  = (__bf16)b1[e];
                b[8 + e]  = (__bf16)b2[e];  b[12 + e] = (__bf16)b3[e];
            }
            c = wmma_bf16(afrag[kc4], b, c);
        }
        __builtin_amdgcn_wave_barrier();    // keep LDS reads before next tile's writes

        // ---- online softmax per query row (scores in c[r], lanes 0..15)
        const int  scol = t * 16 + l15;
        const bool vcol = scol < ST_;
        float ee[4];
#pragma unroll
        for (int r = 0; r < 4; ++r) {
            float sc = vcol ? c[r] * INV_SCALE : -1.0e30f;
            float v = sc;
            v = fmaxf(v, __shfl_xor(v, 1, 32));
            v = fmaxf(v, __shfl_xor(v, 2, 32));
            v = fmaxf(v, __shfl_xor(v, 4, 32));
            v = fmaxf(v, __shfl_xor(v, 8, 32));
            const float tmax  = __shfl(v, 0, 32);       // broadcast to all 32 lanes
            const float m_new = fmaxf(m_run[r], tmax);
            const float corr  = __expf(m_run[r] - m_new);
            ee[r] = __expf(sc - m_new);                 // 0 for padded cols
            float s = ee[r];
            s += __shfl_xor(s, 1, 32);
            s += __shfl_xor(s, 2, 32);
            s += __shfl_xor(s, 4, 32);
            s += __shfl_xor(s, 8, 32);
            const float psum = __shfl(s, 0, 32);
            l_run[r] = l_run[r] * corr + psum;
            m_run[r] = m_new;
#pragma unroll
            for (int j = 0; j < 4; ++j) acc[r][j] *= corr;
        }

        // ---- ctx += p * V, fused with the V concat-copy (fully coalesced rows)
        const int rem  = ST_ - t * 16;
        const int smax = rem < 16 ? rem : 16;
        for (int si = 0; si < smax; ++si) {
            const int s = t * 16 + si;                  // uniform per wave
            const float* srow = (s < SP_)
                ? (vcache + kin_base + (size_t)s * HD_)
                : (vout + (size_t)s * HD_);
            fx4 vv = __builtin_nontemporal_load((const fx4*)(srow + 4 * lane));
            if (s < SP_)
                __builtin_nontemporal_store(vv, (fx4*)(vout + (size_t)s * HD_ + 4 * lane));
#pragma unroll
            for (int r = 0; r < 4; ++r) {
                const float p = __shfl(ee[r], si, 32);
                acc[r][0] += p * vv[0];
                acc[r][1] += p * vv[1];
                acc[r][2] += p * vv[2];
                acc[r][3] += p * vv[3];
            }
        }
    }

    // ---- merge the 4 waves' partials (LSE) and emit un-normalized block partial
    if (lane == 0) {
#pragma unroll
        for (int r = 0; r < 4; ++r) { red_m[wv][r] = m_run[r]; red_l[wv][r] = l_run[r]; }
    }
#pragma unroll
    for (int r = 0; r < 4; ++r)
#pragma unroll
        for (int j = 0; j < 4; ++j)
            red_acc[wv][r][4 * lane + j] = acc[r][j];
    __syncthreads();

    const int hd = threadIdx.x & 127;
#pragma unroll
    for (int r = 0; r < 4; ++r) {
        const float M = fmaxf(fmaxf(red_m[0][r], red_m[1][r]),
                              fmaxf(red_m[2][r], red_m[3][r]));
        float L = 0.0f, C = 0.0f;
#pragma unroll
        for (int w = 0; w < 4; ++w) {
            const float cw = __expf(red_m[w][r] - M);
            L += cw * red_l[w][r];
            C += cw * red_acc[w][r][hd];
        }
        float* pb = part + ((size_t)blockIdx.x * 4 + r) * PART_STRIDE;
        if (threadIdx.x == 0) { pb[0] = M; pb[1] = L; }
        pb[2 + hd] = C;
    }
}

// ---------------------------------------------------------------------------
// Kernel 3: LSE-merge of the NCHUNK partials per (b,h) -> normalized ctx
// ---------------------------------------------------------------------------
__global__ void attn_merge_kernel(const float* __restrict__ part,
                                  float* __restrict__ ctx_ws)
{
    const int bh = blockIdx.x;            // 0..255
    const int hd = threadIdx.x;           // 0..127
#pragma unroll
    for (int r = 0; r < 4; ++r) {
        float M = -1.0e30f;
#pragma unroll
        for (int ck = 0; ck < NCHUNK; ++ck)
            M = fmaxf(M, part[(((size_t)bh * NCHUNK + ck) * 4 + r) * PART_STRIDE]);
        float L = 0.0f, C = 0.0f;
#pragma unroll
        for (int ck = 0; ck < NCHUNK; ++ck) {
            const float* pb = part + (((size_t)bh * NCHUNK + ck) * 4 + r) * PART_STRIDE;
            const float cw = __expf(pb[0] - M);
            L += cw * pb[1];
            C += cw * pb[2 + hd];
        }
        ctx_ws[((size_t)bh * SN_ + r) * HD_ + hd] = C / L;
    }
}

// ---------------------------------------------------------------------------
// Kernel 4: output projection out = merged(ctx) @ Wo^T  (64x2048x2048)
// ---------------------------------------------------------------------------
__global__ void out_proj_kernel(const float* __restrict__ ctx_ws,
                                const float* __restrict__ Wo,
                                float* __restrict__ out)
{
    const int wave = blockIdx.x * (blockDim.x >> 5) + (threadIdx.x >> 5);  // 0..511
    const int lane = threadIdx.x & 31;
    const int half = lane >> 4;
    const int l15  = lane & 15;
    const int mt   = wave >> 7;
    const int nt   = wave & 127;

    v8f c;
#pragma unroll
    for (int i = 0; i < 8; ++i) c[i] = 0.0f;

    const int row = mt * 16 + l15;
    const int b_  = row >> 2, s_ = row & 3;
    const float* wrow = Wo + (size_t)(nt * 16 + l15) * D_;

    for (int k0 = 0; k0 < D_; k0 += 32) {
        const int ka = k0 + 8 * half;            // first 8-run of K
        const int kb = k0 + 16 + 8 * half;       // second 8-run of K
        const float* pa = ctx_ws + (((size_t)(b_ * H_ + (ka >> 7))) * SN_ + s_) * HD_ + (ka & 127);
        const float* pb = ctx_ws + (((size_t)(b_ * H_ + (kb >> 7))) * SN_ + s_) * HD_ + (kb & 127);
        fx4 a0 = *(const fx4*)(pa);
        fx4 a1 = *(const fx4*)(pa + 4);
        fx4 a2 = *(const fx4*)(pb);
        fx4 a3 = *(const fx4*)(pb + 4);
        fx4 b0 = *(const fx4*)(wrow + k0 + 8 * half);
        fx4 b1 = *(const fx4*)(wrow + k0 + 8 * half + 4);
        fx4 b2 = *(const fx4*)(wrow + k0 + 16 + 8 * half);
        fx4 b3 = *(const fx4*)(wrow + k0 + 16 + 8 * half + 4);
        v16bf a, b;
#pragma unroll
        for (int e = 0; e < 4; ++e) {
            a[e]      = (__bf16)a0[e];  a[4 + e]  = (__bf16)a1[e];
            a[8 + e]  = (__bf16)a2[e];  a[12 + e] = (__bf16)a3[e];
            b[e]      = (__bf16)b0[e];  b[4 + e]  = (__bf16)b1[e];
            b[8 + e]  = (__bf16)b2[e];  b[12 + e] = (__bf16)b3[e];
        }
        c = wmma_bf16(a, b, c);
    }

#pragma unroll
    for (int v = 0; v < 8; ++v) {
        const int r = mt * 16 + v + 8 * half;
        const int j = nt * 16 + l15;
        out[(size_t)r * D_ + j] = c[v];
    }
}

// ---------------------------------------------------------------------------
extern "C" void kernel_launch(void* const* d_in, const int* in_sizes, int n_in,
                              void* d_out, int out_size, void* d_ws, size_t ws_size,
                              hipStream_t stream) {
    (void)in_sizes; (void)n_in; (void)out_size; (void)ws_size;
    const float* x  = (const float*)d_in[0];
    const float* kc = (const float*)d_in[1];
    const float* vc = (const float*)d_in[2];
    const float* Wq = (const float*)d_in[3];
    const float* Wk = (const float*)d_in[4];
    const float* Wv = (const float*)d_in[5];
    const float* Wo = (const float*)d_in[6];
    float* out = (float*)d_out;

    float* q_ws   = (float*)d_ws;                              // (B,H,4,128) f32
    float* ctx_ws = q_ws + (size_t)B_ * H_ * SN_ * HD_;        // (B,H,4,128) f32
    float* part   = ctx_ws + (size_t)B_ * H_ * SN_ * HD_;      // 1024 * 4 * 130 f32

    // 1) QKV projections (WMMA bf16); writes q workspace + new K/V cache rows
    qkv_proj_kernel<<<384, 128, 0, stream>>>(x, Wq, Wk, Wv, q_ws, out);
    // 2) Fused KV concat-copy + flash-decode attention partials
    //    (single pass over the 1.07 GB of cached K/V: the roofline-dominant term)
    attn_partial_kernel<<<B_ * H_ * NCHUNK, 128, 0, stream>>>(q_ws, kc, vc, out, part);
    // 3) LSE merge of s-chunk partials -> normalized ctx
    attn_merge_kernel<<<B_ * H_, 128, 0, stream>>>(part, ctx_ws);
    // 4) Output projection (WMMA bf16)
    out_proj_kernel<<<128, 128, 0, stream>>>(ctx_ws, Wo, out);
}
